// TextEncoder_30210799960279
// MI455X (gfx1250) — matrix operand order repaired and checked by reading, not compiled, not run on hardware
//
#include <hip/hip_runtime.h>
#include <hip/hip_bf16.h>

// Problem constants
#define BQ    64     // batch
#define NQ    20     // tokens
#define DQ    512    // embed dim
#define HQ    512    // hidden
#define G4    2048   // 4*H (gates)
#define K2H   1024   // 2*H
#define NSPAN 190    // N*(N-1)/2
#define NTQ   30
#define SEMQ  64
#define OUTC  1920   // NT*SEM

typedef __bf16 bf16_t;
typedef __attribute__((ext_vector_type(16))) __bf16 v16bf;
typedef __attribute__((ext_vector_type(8)))  __bf16 v8bf;
typedef __attribute__((ext_vector_type(8)))  float  v8f;

__device__ __forceinline__ float sigmf(float x) { return 1.0f / (1.0f + expf(-x)); }

// ---------------- float -> bf16 convert ----------------
__global__ void f2bf_kernel(const float* __restrict__ in, bf16_t* __restrict__ out, int n) {
    int i = blockIdx.x * blockDim.x + threadIdx.x;
    if (i < n) out[i] = (bf16_t)in[i];
}

// ---------------- embedding + length mask, token-major [t][b][d] ----------------
__global__ void embed_kernel(const int* __restrict__ caps, const int* __restrict__ lens,
                             const float* __restrict__ emb, bf16_t* __restrict__ wv) {
    int gid = blockIdx.x * blockDim.x + threadIdx.x;
    int d = gid % DQ;
    int b = (gid / DQ) % BQ;
    int t = gid / (DQ * BQ);
    if (t >= NQ) return;
    int tok = caps[b * NQ + t];
    float v = (t < lens[b]) ? emb[(size_t)tok * DQ + d] : 0.0f;
    wv[gid] = (bf16_t)v;
}

// ---- fragment loaders (CDNA5 16x16x32 bf16 layouts) ----
// A 16x32: lane<16 -> row=laneLo, halves K {0..7,16..23}; lane>=16 -> +8
__device__ __forceinline__ void load_afrag(v16bf& f, const bf16_t* p, int kk) {
    v8bf x = *(const v8bf*)(p + kk);
    v8bf y = *(const v8bf*)(p + kk + 16);
    #pragma unroll
    for (int i = 0; i < 8; ++i) { f[i] = x[i]; f[i + 8] = y[i]; }
}
// B 32x16: lane<16 -> col=laneLo, K {0..7,16..23}; lane>=16 -> +8 (p already lane-adjusted)
__device__ __forceinline__ void load_bfrag(v16bf& f, const bf16_t* p, int kk) {
    v8bf x = *(const v8bf*)(p + kk);
    v8bf y = *(const v8bf*)(p + kk + 16);
    #pragma unroll
    for (int i = 0; i < 8; ++i) { f[i] = x[i]; f[i + 8] = y[i]; }
}

// ---------------- bf16 WMMA GEMM:  C[M,Ncols] = A[M,K] * W[Ncols,K]^T (+bias) ----------------
// A row-major bf16, W row-major bf16 (PyTorch weight layout), C f32 with stride ldc.
// One wave -> 32(M) x 64(N) strip (2 M-subtiles x 4 N-subtiles); block = 8 waves = 256 rows.
// K must be a multiple of 64; M a multiple of 32; Ncols a multiple of 64.
__global__ __launch_bounds__(256)
void gemm_bf16_wmma(const bf16_t* __restrict__ A, const bf16_t* __restrict__ W,
                    float* __restrict__ C, int M, int K, int Ncols, int ldc,
                    const float* __restrict__ bias0, const float* __restrict__ bias1) {
    const int wave   = threadIdx.x >> 5;
    const int lane   = threadIdx.x & 31;
    const int laneLo = lane & 15;
    const int laneHi = lane >> 4;

    const int mTile = blockIdx.y * 8 + wave;
    const int row0  = mTile * 32;
    if (row0 >= M) return;
    const int n0 = blockIdx.x * 64;

    v8f acc0[4], acc1[4];
    #pragma unroll
    for (int j = 0; j < 4; ++j) {
        acc0[j] = (v8f){0.f,0.f,0.f,0.f,0.f,0.f,0.f,0.f};
        acc1[j] = (v8f){0.f,0.f,0.f,0.f,0.f,0.f,0.f,0.f};
    }

    const bf16_t* ap0 = A + (size_t)(row0 + laneLo) * K + laneHi * 8;
    const bf16_t* ap1 = ap0 + (size_t)16 * K;
    const bf16_t* wp[4];
    #pragma unroll
    for (int j = 0; j < 4; ++j)
        wp[j] = W + (size_t)(n0 + j * 16 + laneLo) * K + laneHi * 8;

    // ping-pong K pipeline: prefetch K+32 while computing K, K+64 between halves
    v16bf A0[2], B0[4], A1[2], B1[4];
    load_afrag(A0[0], ap0, 0);
    load_afrag(A0[1], ap1, 0);
    #pragma unroll
    for (int j = 0; j < 4; ++j) load_bfrag(B0[j], wp[j], 0);

    for (int kk = 0; kk < K; kk += 64) {
        load_afrag(A1[0], ap0, kk + 32);
        load_afrag(A1[1], ap1, kk + 32);
        #pragma unroll
        for (int j = 0; j < 4; ++j) load_bfrag(B1[j], wp[j], kk + 32);

        #pragma unroll
        for (int j = 0; j < 4; ++j) {
            acc0[j] = __builtin_amdgcn_wmma_f32_16x16x32_bf16(
                false, A0[0], false, B0[j], (short)0, acc0[j], false, false);
            acc1[j] = __builtin_amdgcn_wmma_f32_16x16x32_bf16(
                false, A0[1], false, B0[j], (short)0, acc1[j], false, false);
        }

        if (kk + 64 < K) {
            load_afrag(A0[0], ap0, kk + 64);
            load_afrag(A0[1], ap1, kk + 64);
            #pragma unroll
            for (int j = 0; j < 4; ++j) load_bfrag(B0[j], wp[j], kk + 64);
        }

        #pragma unroll
        for (int j = 0; j < 4; ++j) {
            acc0[j] = __builtin_amdgcn_wmma_f32_16x16x32_bf16(
                false, A1[0], false, B1[j], (short)0, acc0[j], false, false);
            acc1[j] = __builtin_amdgcn_wmma_f32_16x16x32_bf16(
                false, A1[1], false, B1[j], (short)0, acc1[j], false, false);
        }
    }

    // C/D layout: VGPR i -> M = i + 8*laneHi, N = laneLo
    #pragma unroll
    for (int j = 0; j < 4; ++j) {
        int n = n0 + j * 16 + laneLo;
        float badd = 0.0f;
        if (bias0) badd += bias0[n];
        if (bias1) badd += bias1[n];
        #pragma unroll
        for (int i = 0; i < 8; ++i) {
            int r = row0 + laneHi * 8 + i;
            C[(size_t)r * ldc + n]        = acc0[j][i] + badd;
            C[(size_t)(r + 16) * ldc + n] = acc1[j][i] + badd;
        }
    }
}

// ---------------- forward LSTM cell at absolute time t ----------------
// State layout [s][B][H]; gbuf rows = s*B+b (only valid for s<t); xp rows [t][b][4H].
__global__ void fwd_cell_kernel(const float* __restrict__ xp, const float* __restrict__ gbuf,
                                float* __restrict__ cst, bf16_t* __restrict__ hbf,
                                float* __restrict__ ssum, bf16_t* __restrict__ rowbuf_bf, int t) {
    int gid = blockIdx.x * blockDim.x + threadIdx.x;
    int total = (t + 1) * BQ * HQ;
    if (gid >= total) return;
    int u = gid % HQ;
    int b = (gid / HQ) % BQ;
    int s = gid / (HQ * BQ);
    int rs = s * BQ + b;

    const float* xr = xp + (size_t)(t * BQ + b) * G4;
    float pi = xr[u], pf = xr[u + HQ], pg = xr[u + 2 * HQ], po = xr[u + 3 * HQ];
    float cprev = 0.0f, sprev = 0.0f;
    if (s < t) {
        const float* gr = gbuf + (size_t)rs * G4;
        pi += gr[u]; pf += gr[u + HQ]; pg += gr[u + 2 * HQ]; po += gr[u + 3 * HQ];
        cprev = cst[(size_t)rs * HQ + u];
        sprev = ssum[(size_t)rs * HQ + u];
    }
    float c = sigmf(pf) * cprev + sigmf(pi) * tanhf(pg);
    float h = sigmf(po) * tanhf(c);
    cst[(size_t)rs * HQ + u] = c;
    hbf[(size_t)rs * HQ + u] = (bf16_t)h;
    float S = sprev + h;
    ssum[(size_t)rs * HQ + u] = S;
    if (t > s) {  // span (s, e=t), length k+1, k = t-s
        int k = t - s;
        int span = (k - 1) * NQ - (k * (k - 1)) / 2 + s;
        rowbuf_bf[((size_t)b * NSPAN + span) * K2H + u] = (bf16_t)S;
    }
}

// ---------------- backward LSTM cell at absolute time t (runs t = N-1 .. 0) ----------------
// State layout [e][B][H]; gbuf rows = (e-(t+1))*B+b (valid for e>t).
__global__ void bwd_cell_kernel(const float* __restrict__ xp, const float* __restrict__ gbuf,
                                float* __restrict__ cst, bf16_t* __restrict__ hbf,
                                float* __restrict__ ssum, bf16_t* __restrict__ rowbuf_bf, int t) {
    int gid = blockIdx.x * blockDim.x + threadIdx.x;
    int total = (NQ - t) * BQ * HQ;
    if (gid >= total) return;
    int u = gid % HQ;
    int b = (gid / HQ) % BQ;
    int e = t + gid / (HQ * BQ);
    int rs = e * BQ + b;

    const float* xr = xp + (size_t)(t * BQ + b) * G4;
    float pi = xr[u], pf = xr[u + HQ], pg = xr[u + 2 * HQ], po = xr[u + 3 * HQ];
    float cprev = 0.0f, sprev = 0.0f;
    if (e > t) {
        const float* gr = gbuf + (size_t)((e - (t + 1)) * BQ + b) * G4;
        pi += gr[u]; pf += gr[u + HQ]; pg += gr[u + 2 * HQ]; po += gr[u + 3 * HQ];
        cprev = cst[(size_t)rs * HQ + u];
        sprev = ssum[(size_t)rs * HQ + u];
    }
    float c = sigmf(pf) * cprev + sigmf(pi) * tanhf(pg);
    float h = sigmf(po) * tanhf(c);
    cst[(size_t)rs * HQ + u] = c;
    hbf[(size_t)rs * HQ + u] = (bf16_t)h;
    float S = sprev + h;
    ssum[(size_t)rs * HQ + u] = S;
    if (e > t) {  // span (s=t, e), k = e-t; backward half goes in cols [512,1024)
        int k = e - t;
        int span = (k - 1) * NQ - (k * (k - 1)) / 2 + t;
        rowbuf_bf[((size_t)b * NSPAN + span) * K2H + HQ + u] = (bf16_t)S;
    }
}

// ---------------- bias (len * b_out) + L2 normalize each [SEM]=64 chunk, in place ----------------
__global__ __launch_bounds__(256)
void norm_kernel(float* __restrict__ out, const float* __restrict__ b_out) {
    int wave = threadIdx.x >> 5;
    int lane = threadIdx.x & 31;
    int g = blockIdx.x * 8 + wave;
    int total = BQ * NSPAN * NTQ;
    if (g >= total) return;
    int nt = g % NTQ;
    int r  = g / NTQ;              // r = b*NSPAN + span
    int span = r % NSPAN;

    // recover span length: off(k) = (k-1)*N - k(k-1)/2, sizes N-k
    int k = 1, offn = 0;
    while (span >= offn + (NQ - k)) { offn += (NQ - k); ++k; }
    int len = k + 1;

    float* p = out + (size_t)r * OUTC + nt * SEMQ;
    const float* bb = b_out + nt * SEMQ;
    float v0 = p[lane]      + (float)len * bb[lane];
    float v1 = p[lane + 32] + (float)len * bb[lane + 32];
    float ss = v0 * v0 + v1 * v1;
    #pragma unroll
    for (int m = 16; m > 0; m >>= 1) ss += __shfl_xor(ss, m, 32);
    float nrm = fmaxf(sqrtf(ss), 1e-12f);
    float inv = 1.0f / nrm;
    p[lane]      = v0 * inv;
    p[lane + 32] = v1 * inv;
}

// =====================================================================================
extern "C" void kernel_launch(void* const* d_in, const int* in_sizes, int n_in,
                              void* d_out, int out_size, void* d_ws, size_t ws_size,
                              hipStream_t stream) {
    const int*   caps   = (const int*)  d_in[0];
    const int*   lens   = (const int*)  d_in[1];
    const float* emb    = (const float*)d_in[2];
    const float* w_ih_f = (const float*)d_in[3];
    const float* w_hh_f = (const float*)d_in[4];
    const float* b_ih_f = (const float*)d_in[5];
    const float* b_hh_f = (const float*)d_in[6];
    const float* w_ih_b = (const float*)d_in[7];
    const float* w_hh_b = (const float*)d_in[8];
    const float* b_ih_b = (const float*)d_in[9];
    const float* b_hh_b = (const float*)d_in[10];
    const float* w_out  = (const float*)d_in[11];
    const float* b_out  = (const float*)d_in[12];
    float* outp = (float*)d_out;

    // ---- workspace carve-up (256B aligned) ----
    char* ws = (char*)d_ws;
    size_t off = 0;
    auto carve = [&](size_t bytes) { char* p = ws + off; off += (bytes + 255) & ~(size_t)255; return p; };
    bf16_t* wv_bf     = (bf16_t*)carve((size_t)NQ * BQ * DQ * 2);
    bf16_t* w_ih_f_bf = (bf16_t*)carve((size_t)G4 * DQ * 2);
    bf16_t* w_hh_f_bf = (bf16_t*)carve((size_t)G4 * HQ * 2);
    bf16_t* w_ih_b_bf = (bf16_t*)carve((size_t)G4 * DQ * 2);
    bf16_t* w_hh_b_bf = (bf16_t*)carve((size_t)G4 * HQ * 2);
    bf16_t* w_out_bf  = (bf16_t*)carve((size_t)OUTC * K2H * 2);
    float*  xp_f      = (float*) carve((size_t)NQ * BQ * G4 * 4);
    float*  xp_b      = (float*) carve((size_t)NQ * BQ * G4 * 4);
    float*  gbuf      = (float*) carve((size_t)NQ * BQ * G4 * 4);   // >= 1216 rows
    bf16_t* hbf_f     = (bf16_t*)carve((size_t)NQ * BQ * HQ * 2);
    bf16_t* hbf_b     = (bf16_t*)carve((size_t)NQ * BQ * HQ * 2);
    float*  cst_f     = (float*) carve((size_t)NQ * BQ * HQ * 4);
    float*  cst_b     = (float*) carve((size_t)NQ * BQ * HQ * 4);
    float*  ssum_f    = (float*) carve((size_t)NQ * BQ * HQ * 4);
    float*  ssum_b    = (float*) carve((size_t)NQ * BQ * HQ * 4);
    bf16_t* rowbuf_bf = (bf16_t*)carve((size_t)BQ * NSPAN * K2H * 2);
    (void)ws_size; (void)in_sizes; (void)n_in; (void)out_size;

    auto conv = [&](const float* src, bf16_t* dst, int n) {
        f2bf_kernel<<<(n + 255) / 256, 256, 0, stream>>>(src, dst, n);
    };
    auto gemm = [&](const bf16_t* A, const bf16_t* W, float* C, int M, int K, int Ncols,
                    int ldc, const float* bias0, const float* bias1) {
        dim3 grid(Ncols / 64, (M + 255) / 256);   // 8 waves x 32 rows per block
        gemm_bf16_wmma<<<grid, 256, 0, stream>>>(A, W, C, M, K, Ncols, ldc, bias0, bias1);
    };

    // 1) weights -> bf16
    conv(w_ih_f, w_ih_f_bf, G4 * DQ);
    conv(w_hh_f, w_hh_f_bf, G4 * HQ);
    conv(w_ih_b, w_ih_b_bf, G4 * DQ);
    conv(w_hh_b, w_hh_b_bf, G4 * HQ);
    conv(w_out,  w_out_bf,  OUTC * K2H);

    // 2) embed + mask -> token-major bf16 [t][b][d]
    embed_kernel<<<(NQ * BQ * DQ + 255) / 256, 256, 0, stream>>>(caps, lens, emb, wv_bf);

    // 3) input projections for all tokens: xp = wv @ W_ih^T + (b_ih + b_hh)
    gemm(wv_bf, w_ih_f_bf, xp_f, NQ * BQ, DQ, G4, G4, b_ih_f, b_hh_f);
    gemm(wv_bf, w_ih_b_bf, xp_b, NQ * BQ, DQ, G4, G4, b_ih_b, b_hh_b);

    // 4) forward recurrence: one LSTM per start s, batched across (s,b)
    for (int t = 0; t < NQ; ++t) {
        int m = BQ * t;  // rows with s<t need h@W_hh^T
        if (m > 0) gemm(hbf_f, w_hh_f_bf, gbuf, m, HQ, G4, G4, nullptr, nullptr);
        int threads = (t + 1) * BQ * HQ;
        fwd_cell_kernel<<<(threads + 255) / 256, 256, 0, stream>>>(
            xp_f, gbuf, cst_f, hbf_f, ssum_f, rowbuf_bf, t);
    }

    // 5) backward recurrence: one LSTM per end e, t runs high->low
    for (int t = NQ - 1; t >= 0; --t) {
        int m = BQ * (NQ - 1 - t);  // rows with e>t
        if (m > 0) gemm(hbf_b + (size_t)(t + 1) * BQ * HQ, w_hh_b_bf, gbuf, m, HQ, G4, G4,
                        nullptr, nullptr);
        int threads = (NQ - t) * BQ * HQ;
        bwd_cell_kernel<<<(threads + 255) / 256, 256, 0, stream>>>(
            xp_b, gbuf, cst_b, hbf_b, ssum_b, rowbuf_bf, t);
    }

    // 6) output projection: [B*190, 1024] x [1024, 1920] -> d_out (raw, bias/norm fused next)
    gemm(rowbuf_bf, w_out_bf, outp, BQ * NSPAN, K2H, OUTC, OUTC, nullptr, nullptr);

    // 7) add len*b_out and L2-normalize each 64-vector, in place
    int groups = BQ * NSPAN * NTQ;
    norm_kernel<<<(groups + 7) / 8, 256, 0, stream>>>(outp, b_out);
}
